// LSTM1_30039001268551
// MI455X (gfx1250) — matrix operand order repaired and checked by reading, not compiled, not run on hardware
//
#include <hip/hip_runtime.h>

typedef _Float16 half16 __attribute__((ext_vector_type(16)));
typedef float    float8 __attribute__((ext_vector_type(8)));

static constexpr int kSL   = 10;    // seq len
static constexpr int kHID  = 20;    // hidden
static constexpr int kIN   = 128;   // input size
static constexpr int kSEQB = 16;    // sequences per block
static constexpr int kROWS = kSEQB * kSL; // 160 flattened (b,t) rows per block
static constexpr int kNG   = 4 * kHID;    // 80 gate columns per direction

__device__ __forceinline__ float sigmf(float v) { return 1.0f / (1.0f + __expf(-v)); }
__device__ __forceinline__ float tanh_fast(float v) {
  float e = __expf(2.0f * v);
  return (e - 1.0f) / (e + 1.0f);
}

// A fragment (16x32 f16, M x K). Per CDNA5 ISA: lane<16 holds row M=lane,
// K in {kb..kb+7} U {kb+16..kb+23} with kb = s*32; lane>=16 same row, K shifted by +8.
__device__ __forceinline__ half16 loadA(const float* __restrict__ rowp, int s, int lane) {
  const int kb = s * 32 + ((lane >> 4) << 3);
  const float4 v0 = *(const float4*)(rowp + kb);
  const float4 v1 = *(const float4*)(rowp + kb + 4);
  const float4 v2 = *(const float4*)(rowp + kb + 16);
  const float4 v3 = *(const float4*)(rowp + kb + 20);
  half16 a;
  a[0] = (_Float16)v0.x;  a[1] = (_Float16)v0.y;  a[2] = (_Float16)v0.z;  a[3] = (_Float16)v0.w;
  a[4] = (_Float16)v1.x;  a[5] = (_Float16)v1.y;  a[6] = (_Float16)v1.z;  a[7] = (_Float16)v1.w;
  a[8] = (_Float16)v2.x;  a[9] = (_Float16)v2.y;  a[10] = (_Float16)v2.z; a[11] = (_Float16)v2.w;
  a[12] = (_Float16)v3.x; a[13] = (_Float16)v3.y; a[14] = (_Float16)v3.z; a[15] = (_Float16)v3.w;
  return a;
}

// B fragment (32x16 f16, K x N). Same K order as A: lane<16 holds col N=lane,
// K in {kb..kb+7} U {kb+16..kb+23}; lane>=16 K shifted by +8. B[k][n] = W[n][k].
__device__ __forceinline__ half16 loadB(const float* __restrict__ rowp, int s, int lane) {
  const int kb = s * 32 + ((lane >> 4) << 3);
  const float4 v0 = *(const float4*)(rowp + kb);
  const float4 v1 = *(const float4*)(rowp + kb + 4);
  const float4 v2 = *(const float4*)(rowp + kb + 16);
  const float4 v3 = *(const float4*)(rowp + kb + 20);
  half16 b;
  b[0] = (_Float16)v0.x;  b[1] = (_Float16)v0.y;  b[2] = (_Float16)v0.z;  b[3] = (_Float16)v0.w;
  b[4] = (_Float16)v1.x;  b[5] = (_Float16)v1.y;  b[6] = (_Float16)v1.z;  b[7] = (_Float16)v1.w;
  b[8] = (_Float16)v2.x;  b[9] = (_Float16)v2.y;  b[10] = (_Float16)v2.z; b[11] = (_Float16)v2.w;
  b[12] = (_Float16)v3.x; b[13] = (_Float16)v3.y; b[14] = (_Float16)v3.z; b[15] = (_Float16)v3.w;
  return b;
}

__global__ __launch_bounds__(320)
void lstm_bidir_fused(const float* __restrict__ x,
                      const float* __restrict__ WihF, const float* __restrict__ WhhF,
                      const float* __restrict__ bihF, const float* __restrict__ bhhF,
                      const float* __restrict__ WihB,
                      const float* __restrict__ bihB, const float* __restrict__ bhhB,
                      float* __restrict__ out) {
  __shared__ float xpF[kROWS][kNG];      // fwd input projections (+bias), fp32
  __shared__ float xpB[kSEQB][kNG];      // bwd projections for t=SL-1 only
  __shared__ float hF[kSEQB][kHID];      // forward hidden state
  __shared__ float WhhT[kHID][kNG];      // W_hh_f transposed: [k][gate*20+u]

  const int tid  = threadIdx.x;
  const int lane = tid & 31;
  const int wave = tid >> 5;

  // Stage W_hh_f transposed into LDS; zero-init h.
  for (int idx = tid; idx < kNG * kHID; idx += blockDim.x) {
    const int row = idx / kHID, k = idx % kHID;
    WhhT[k][row] = WhhF[idx];
  }
  hF[tid / kHID][tid % kHID] = 0.0f;     // blockDim == 320 == 16*20

  const float* xblk = x + (size_t)blockIdx.x * kROWS * kIN;
  const int mrow = lane & 15;

  // ---------------- Phase A: input projections via WMMA ----------------
  {
    // Forward: wave w owns M-tile w (rows w*16 .. w*16+15). Load A once, reuse 5x.
    half16 aF[4];
#pragma unroll
    for (int s = 0; s < 4; ++s)
      aF[s] = loadA(xblk + (size_t)(wave * 16 + mrow) * kIN, s, lane);

#pragma unroll
    for (int nt = 0; nt < 5; ++nt) {
      const int n = nt * 16 + mrow;            // gate column this lane owns (N=lane&15)
      const float cb = bihF[n] + bhhF[n];      // bias depends only on N -> splat into C
      float8 acc = {cb, cb, cb, cb, cb, cb, cb, cb};
#pragma unroll
      for (int s = 0; s < 4; ++s) {
        half16 bf = loadB(WihF + (size_t)n * kIN, s, lane);
        acc = __builtin_amdgcn_wmma_f32_16x16x32_f16(false, aF[s], false, bf,
                                                     (short)0, acc, false, false);
      }
#pragma unroll
      for (int r = 0; r < 8; ++r) {
        const int ml = (lane < 16) ? r : (r + 8);   // C/D layout: VGPR r -> M row
        xpF[wave * 16 + ml][nt * 16 + mrow] = acc[r];
      }
    }

    // Backward: hs_b[0] only needs x at original t = SL-1 (h0 = c0 = 0).
    // Gather the 16 t=9 rows as one M-tile; waves 0..4 each do one N-tile.
    if (wave < 5) {
      half16 aB[4];
#pragma unroll
      for (int s = 0; s < 4; ++s)
        aB[s] = loadA(xblk + (size_t)(mrow * kSL + (kSL - 1)) * kIN, s, lane);
      const int n = wave * 16 + mrow;
      const float cb = bihB[n] + bhhB[n];
      float8 acc = {cb, cb, cb, cb, cb, cb, cb, cb};
#pragma unroll
      for (int s = 0; s < 4; ++s) {
        half16 bf = loadB(WihB + (size_t)n * kIN, s, lane);
        acc = __builtin_amdgcn_wmma_f32_16x16x32_f16(false, aB[s], false, bf,
                                                     (short)0, acc, false, false);
      }
#pragma unroll
      for (int r = 0; r < 8; ++r) {
        const int ml = (lane < 16) ? r : (r + 8);
        xpB[ml][wave * 16 + mrow] = acc[r];
      }
    }
  }
  __syncthreads();

  // ---------------- Phase B: fp32 recurrence ----------------
  const int bb = tid / kHID;   // sequence within block, 0..15
  const int u  = tid % kHID;   // hidden unit, 0..19
  float cc = 0.0f, hn = 0.0f;

  for (int t = 0; t < kSL; ++t) {
    float g0 = xpF[bb * kSL + t][0 * kHID + u];
    float g1 = xpF[bb * kSL + t][1 * kHID + u];
    float g2 = xpF[bb * kSL + t][2 * kHID + u];
    float g3 = xpF[bb * kSL + t][3 * kHID + u];
#pragma unroll
    for (int k = 0; k < kHID; ++k) {
      const float hk = hF[bb][k];
      g0 += hk * WhhT[k][0 * kHID + u];
      g1 += hk * WhhT[k][1 * kHID + u];
      g2 += hk * WhhT[k][2 * kHID + u];
      g3 += hk * WhhT[k][3 * kHID + u];
    }
    const float iv = sigmf(g0), fv = sigmf(g1);
    const float gv = tanh_fast(g2), ov = sigmf(g3);
    cc = fv * cc + iv * gv;
    hn = ov * tanh_fast(cc);
    __syncthreads();            // all reads of hF done
    hF[bb][u] = hn;
    __syncthreads();            // new h visible
  }

  const size_t ob = ((size_t)blockIdx.x * kSEQB + bb) * (size_t)(2 * kHID);
  out[ob + u] = hn;             // hs_f[-1]

  // Backward single step from zero state: c = i*g, h = o*tanh(c). (f gate is moot.)
  const float ib  = sigmf(xpB[bb][0 * kHID + u]);
  const float gb  = tanh_fast(xpB[bb][2 * kHID + u]);
  const float obv = sigmf(xpB[bb][3 * kHID + u]);
  out[ob + kHID + u] = obv * tanh_fast(ib * gb);
}

extern "C" void kernel_launch(void* const* d_in, const int* in_sizes, int n_in,
                              void* d_out, int out_size, void* d_ws, size_t ws_size,
                              hipStream_t stream) {
  const float* x    = (const float*)d_in[0];
  const float* WihF = (const float*)d_in[1];
  const float* WhhF = (const float*)d_in[2];
  const float* bihF = (const float*)d_in[3];
  const float* bhhF = (const float*)d_in[4];
  const float* WihB = (const float*)d_in[5];
  // d_in[6] = W_hh_b: provably unused (hs_b[0] has h0 = 0)
  const float* bihB = (const float*)d_in[7];
  const float* bhhB = (const float*)d_in[8];
  float* out = (float*)d_out;

  const int rows   = in_sizes[0] / kIN;   // B * SL
  const int seqs   = rows / kSL;          // B
  const int blocks = seqs / kSEQB;        // 2048 for B = 32768

  hipLaunchKernelGGL(lstm_bidir_fused, dim3(blocks), dim3(320), 0, stream,
                     x, WihF, WhhF, bihF, bhhF, WihB, bihB, bhhB, out);
}
